// MaskedCausalAttention_20280835571722
// MI455X (gfx1250) — matrix operand order repaired and checked by reading, not compiled, not run on hardware
//
#include <hip/hip_runtime.h>

#define BATCH 4
#define SEQ   2048
#define HID   512
#define NH    8
#define HD    64

typedef __attribute__((ext_vector_type(16))) __bf16 v16bf;
typedef __attribute__((ext_vector_type(8)))  float  v8f;

union ABOp { v16bf v; unsigned int u[8]; };

__device__ __forceinline__ unsigned int pk_bf16(float a, float b) {
    __bf16 x = (__bf16)a, y = (__bf16)b;
    unsigned int lo = (unsigned int)__builtin_bit_cast(unsigned short, x);
    unsigned int hi = (unsigned int)__builtin_bit_cast(unsigned short, y);
    return lo | (hi << 16);
}

__device__ __forceinline__ unsigned short f2bf(float a) {
    __bf16 x = (__bf16)a;
    return __builtin_bit_cast(unsigned short, x);
}

// 16-bit A operand (16x32): VGPR v, lane-half h -> K index (even, pair k,k+1)
__device__ __forceinline__ int ka_map(int v, int h) {
    return (v < 4) ? (2 * v + 8 * h) : (16 + 2 * (v - 4) + 8 * h);
}
// 16-bit B operand (32x16): VGPR v, lane-half h -> K index (even, pair k,k+1)
__device__ __forceinline__ int kb_map(int v, int h) { return (v < 4) ? (2 * v + 8 * h) : (16 + 2 * (v - 4) + 8 * h); }

// ---------------------------------------------------------------------------
// Activation convert: x f32 -> bf16 (pairs), fully coalesced
// ---------------------------------------------------------------------------
__global__ void x_convert_kernel(const float* __restrict__ X,
                                 unsigned int* __restrict__ XB) {
    int idx = blockIdx.x * blockDim.x + threadIdx.x;  // pair index
    if (idx >= (BATCH * SEQ * HID) / 2) return;
    float2 f = ((const float2*)X)[idx];
    XB[idx] = pk_bf16(f.x, f.y);
}

// ---------------------------------------------------------------------------
// Weight convert + transpose: W[k,n] f32 -> WT[n,k] bf16
// ---------------------------------------------------------------------------
__global__ void wt_convert_kernel(const float* __restrict__ W,
                                  unsigned short* __restrict__ WT) {
    int idx = blockIdx.x * blockDim.x + threadIdx.x;
    if (idx >= HID * HID) return;
    int k = idx >> 9;      // row of W
    int n = idx & 511;     // col of W
    WT[(size_t)n * HID + k] = f2bf(W[idx]);
}

// ---------------------------------------------------------------------------
// GEMM: C(rows x 512) = A(bf16, rows x 512) * W(bf16, via WT[n,k]) + bias
// mode 0: RoPE epilogue, write bf16 row-major                (q, k)
// mode 1: plain epilogue, write bf16 transposed per head      (v -> (b,h,d,t))
// mode 2: plain epilogue, write f32 row-major                 (final projection)
// block = 128 threads (4 waves); each wave: 16x64 tile; workgroup: 64x64
// ---------------------------------------------------------------------------
__global__ __launch_bounds__(128) void gemm_epilogue_kernel(
    const unsigned short* __restrict__ A, const unsigned short* __restrict__ WT,
    const float* __restrict__ bias, void* __restrict__ out, int mode) {
    const int lane = threadIdx.x & 31;
    const int wave = threadIdx.x >> 5;
    const int half = lane >> 4;
    const int l16  = lane & 15;
    const int c0 = blockIdx.x * 64;
    const int m0 = blockIdx.y * 64 + wave * 16;

    v8f acc[4] = {};
    for (int k0 = 0; k0 < HID; k0 += 32) {
        ABOp a;
        const size_t abase = (size_t)(m0 + l16) * HID + k0;
#pragma unroll
        for (int v = 0; v < 8; ++v)
            a.u[v] = *(const unsigned int*)&A[abase + ka_map(v, half)];
#pragma unroll
        for (int j = 0; j < 4; ++j) {
            ABOp bm;
            int n = c0 + j * 16 + l16;
#pragma unroll
            for (int v = 0; v < 8; ++v) {
                int kp = k0 + kb_map(v, half);
                bm.u[v] = *(const unsigned int*)&WT[(size_t)n * HID + kp];
            }
            acc[j] = __builtin_amdgcn_wmma_f32_16x16x32_bf16(
                false, a.v, false, bm.v, (short)0, acc[j], false, false);
        }
    }

    if (mode == 0) {  // bias + RoPE -> bf16 row-major
        unsigned short* O = (unsigned short*)out;
#pragma unroll
        for (int j = 0; j < 4; ++j) {
            int c = c0 + j * 16 + l16;
            float bsv = bias[c];
            int ce = c & ~1;
            float invf = __expf(-9.210340372f * (float)ce * (1.0f / (float)HID));
#pragma unroll
            for (int r = 0; r < 8; ++r) {
                int row = m0 + r + 8 * half;
                int t = row & (SEQ - 1);
                float val = acc[j][r] + bsv;
                float other = __shfl_xor(val, 1, 32);  // RoPE pair partner
                float sn, cs;
                __sincosf((float)t * invf, &sn, &cs);
                float res = (c & 1) ? (other * sn + val * cs)
                                    : (val * cs - other * sn);
                O[(size_t)row * HID + c] = f2bf(res);
            }
        }
    } else if (mode == 1) {  // bias -> bf16, transposed per head (b,h,d,t)
        unsigned short* O = (unsigned short*)out;
#pragma unroll
        for (int j = 0; j < 4; ++j) {
            int c = c0 + j * 16 + l16;
            float bsv = bias[c];
            int h = c >> 6, d = c & 63;
#pragma unroll
            for (int r = 0; r < 8; ++r) {
                int row = m0 + r + 8 * half;
                int b = row / SEQ;
                int t = row & (SEQ - 1);
                float val = acc[j][r] + bsv;
                O[((size_t)(b * NH + h) * HD + d) * SEQ + t] = f2bf(val);
            }
        }
    } else {  // bias -> f32 row-major (d_out)
        float* O = (float*)out;
#pragma unroll
        for (int j = 0; j < 4; ++j) {
            int c = c0 + j * 16 + l16;
            float bsv = bias[c];
#pragma unroll
            for (int r = 0; r < 8; ++r) {
                int row = m0 + r + 8 * half;
                O[(size_t)row * HID + c] = acc[j][r] + bsv;
            }
        }
    }
}

// ---------------------------------------------------------------------------
// Flash attention with causal + mgdt column masking.
// grid.x = B * NH * (SEQ/64); block = 128 (4 waves, each owns 16 query rows)
// Output written directly as bf16 (B,T,H) for the final projection GEMM.
// ---------------------------------------------------------------------------
__global__ __launch_bounds__(128) void attn_kernel(
    const unsigned short* __restrict__ QB, const unsigned short* __restrict__ KB,
    const unsigned short* __restrict__ VT, unsigned short* __restrict__ OUT) {
    __shared__ unsigned short lds_p[4][16 * 64];  // per-wave P staging (C->A relayout)

    const int lane = threadIdx.x & 31;
    const int wave = threadIdx.x >> 5;
    const int half = lane >> 4;
    const int l16  = lane & 15;

    const int q64 = blockIdx.x % (SEQ / 64);
    const int h   = (blockIdx.x / (SEQ / 64)) % NH;
    const int b   = blockIdx.x / ((SEQ / 64) * NH);
    const int t0  = q64 * 64 + wave * 16;

    // Preload Q A-operands for d in [0,32) and [32,64)
    ABOp qa[2];
#pragma unroll
    for (int ks = 0; ks < 2; ++ks)
#pragma unroll
        for (int v = 0; v < 8; ++v) {
            int d = ks * 32 + ka_map(v, half);
            qa[ks].u[v] = *(const unsigned int*)
                &QB[(size_t)(b * SEQ + t0 + l16) * HID + h * HD + d];
        }

    v8f o[4] = {};
    float mrun[8], lrun[8];
#pragma unroll
    for (int r = 0; r < 8; ++r) { mrun[r] = -1e30f; lrun[r] = 0.0f; }

    const int nblk = q64 + 1;
    for (int blk = 0; blk < nblk; ++blk) {
        const int s0 = blk * 64;

        // Prefetch next K/V block into cache (global_prefetch_b8)
        if (blk + 1 < nblk) {
            const int sn = s0 + 64;
            __builtin_prefetch(
                &KB[(size_t)(b * SEQ + sn + lane) * HID + h * HD], 0, 0);
            __builtin_prefetch(
                &KB[(size_t)(b * SEQ + sn + 32 + lane) * HID + h * HD], 0, 0);
            __builtin_prefetch(
                &VT[((size_t)(b * NH + h) * HD + lane) * SEQ + sn], 0, 0);
            __builtin_prefetch(
                &VT[((size_t)(b * NH + h) * HD + 32 + lane) * SEQ + sn], 0, 0);
        }

        v8f sc[4];
#pragma unroll
        for (int j = 0; j < 4; ++j) {
            v8f cacc = {};
            const int sg = s0 + j * 16 + l16;  // key column this lane owns
#pragma unroll
            for (int ks = 0; ks < 2; ++ks) {
                ABOp bm;
#pragma unroll
                for (int v = 0; v < 8; ++v) {
                    int d = ks * 32 + kb_map(v, half);
                    bm.u[v] = *(const unsigned int*)
                        &KB[(size_t)(b * SEQ + sg) * HID + h * HD + d];
                }
                cacc = __builtin_amdgcn_wmma_f32_16x16x32_bf16(
                    false, qa[ks].v, false, bm.v, (short)0, cacc, false, false);
            }
            const bool colmask = (sg >= 4) && ((sg % 3) == 1);
#pragma unroll
            for (int r = 0; r < 8; ++r) {
                int tg = t0 + r + 8 * half;
                float s = cacc[r] * 0.125f;  // 1/sqrt(64)
                cacc[r] = (sg > tg || colmask) ? -1e30f : s;
            }
            sc[j] = cacc;
        }

        // Online softmax row statistics (rows m=r+8*half, reduce over 16 lanes)
        float fac[8];
#pragma unroll
        for (int r = 0; r < 8; ++r) {
            float ml = fmaxf(fmaxf(sc[0][r], sc[1][r]), fmaxf(sc[2][r], sc[3][r]));
#pragma unroll
            for (int mk = 1; mk < 16; mk <<= 1) ml = fmaxf(ml, __shfl_xor(ml, mk, 32));
            float mn = fmaxf(mrun[r], ml);
            fac[r] = __expf(mrun[r] - mn);
            mrun[r] = mn;
            float ls = 0.0f;
#pragma unroll
            for (int j = 0; j < 4; ++j) {
                float p = __expf(sc[j][r] - mn);
                sc[j][r] = p;
                ls += p;
            }
#pragma unroll
            for (int mk = 1; mk < 16; mk <<= 1) ls += __shfl_xor(ls, mk, 32);
            lrun[r] = lrun[r] * fac[r] + ls;
        }
#pragma unroll
        for (int j2 = 0; j2 < 4; ++j2)
#pragma unroll
            for (int r = 0; r < 8; ++r) o[j2][r] *= fac[r];

        // P: C layout -> LDS (m,s) -> A layout (same-wave DS ops are in-order)
#pragma unroll
        for (int j = 0; j < 4; ++j)
#pragma unroll
            for (int r = 0; r < 8; ++r)
                lds_p[wave][(r + 8 * half) * 64 + j * 16 + l16] = f2bf(sc[j][r]);
        __builtin_amdgcn_wave_barrier();
        ABOp pa[2];
#pragma unroll
        for (int ks2 = 0; ks2 < 2; ++ks2)
#pragma unroll
            for (int v = 0; v < 8; ++v) {
                int idx = l16 * 64 + ks2 * 32 + ka_map(v, half);
                pa[ks2].u[v] = *(const unsigned int*)&lds_p[wave][idx];
            }
        __builtin_amdgcn_wave_barrier();

        // o += P @ V  (V stored transposed per head: (b,h,d,t))
#pragma unroll
        for (int j2 = 0; j2 < 4; ++j2) {
            const int d = j2 * 16 + l16;
#pragma unroll
            for (int ks2 = 0; ks2 < 2; ++ks2) {
                ABOp bm;
#pragma unroll
                for (int v = 0; v < 8; ++v) {
                    int sp = s0 + ks2 * 32 + kb_map(v, half);
                    bm.u[v] = *(const unsigned int*)
                        &VT[((size_t)(b * NH + h) * HD + d) * SEQ + sp];
                }
                o[j2] = __builtin_amdgcn_wmma_f32_16x16x32_bf16(
                    false, pa[ks2].v, false, bm.v, (short)0, o[j2], false, false);
            }
        }
    }

    // Normalize and store bf16 attention output (B,T,H) for final projection
#pragma unroll
    for (int j2 = 0; j2 < 4; ++j2) {
        const int d = j2 * 16 + l16;
#pragma unroll
        for (int r = 0; r < 8; ++r) {
            int tg = t0 + r + 8 * half;
            OUT[(size_t)(b * SEQ + tg) * HID + h * HD + d] =
                f2bf(o[j2][r] * (1.0f / lrun[r]));
        }
    }
}

// ---------------------------------------------------------------------------
extern "C" void kernel_launch(void* const* d_in, const int* in_sizes, int n_in,
                              void* d_out, int out_size, void* d_ws, size_t ws_size,
                              hipStream_t stream) {
    (void)in_sizes; (void)n_in; (void)out_size; (void)ws_size;
    const float* x  = (const float*)d_in[0];
    const float* Wq = (const float*)d_in[1];
    const float* bq = (const float*)d_in[2];
    const float* Wk = (const float*)d_in[3];
    const float* bk = (const float*)d_in[4];
    const float* Wv = (const float*)d_in[5];
    const float* bv = (const float*)d_in[6];
    const float* Wo = (const float*)d_in[7];
    const float* bo = (const float*)d_in[8];

    char* ws = (char*)d_ws;
    const size_t SZbf = (size_t)BATCH * SEQ * HID * 2;  // 8 MB per bf16 tensor
    unsigned short* xb   = (unsigned short*)(ws);
    unsigned short* qb   = (unsigned short*)(ws + SZbf);
    unsigned short* kb   = (unsigned short*)(ws + 2 * SZbf);
    unsigned short* vT   = (unsigned short*)(ws + 3 * SZbf);
    unsigned short* attnb= (unsigned short*)(ws + 4 * SZbf);
    unsigned short* wTq  = (unsigned short*)(ws + 5 * SZbf);
    unsigned short* wTk  = wTq + (size_t)HID * HID;
    unsigned short* wTv  = wTk + (size_t)HID * HID;
    unsigned short* wTo  = wTv + (size_t)HID * HID;

    // Converts
    {
        int npair = (BATCH * SEQ * HID) / 2;
        x_convert_kernel<<<dim3((npair + 255) / 256), dim3(256), 0, stream>>>(
            x, (unsigned int*)xb);
    }
    dim3 bW(256), gW((HID * HID + 255) / 256);
    wt_convert_kernel<<<gW, bW, 0, stream>>>(Wq, wTq);
    wt_convert_kernel<<<gW, bW, 0, stream>>>(Wk, wTk);
    wt_convert_kernel<<<gW, bW, 0, stream>>>(Wv, wTv);
    wt_convert_kernel<<<gW, bW, 0, stream>>>(Wo, wTo);

    // QKV projections (+bias, +RoPE for q/k, V transposed per head)
    dim3 gg(HID / 64, (BATCH * SEQ) / 64), bb(128);
    gemm_epilogue_kernel<<<gg, bb, 0, stream>>>(xb, wTq, bq, qb, 0);
    gemm_epilogue_kernel<<<gg, bb, 0, stream>>>(xb, wTk, bk, kb, 0);
    gemm_epilogue_kernel<<<gg, bb, 0, stream>>>(xb, wTv, bv, vT, 1);

    // Flash attention
    attn_kernel<<<dim3(BATCH * NH * (SEQ / 64)), dim3(128), 0, stream>>>(
        qb, kb, vT, attnb);

    // Final projection -> f32 d_out
    gemm_epilogue_kernel<<<gg, bb, 0, stream>>>(attnb, wTo, bo, d_out, 2);
}
